// SparseConv_C_90555090469376
// MI455X (gfx1250) — matrix operand order repaired and checked by reading, not compiled, not run on hardware
//
#include <hip/hip_runtime.h>

typedef __attribute__((ext_vector_type(16))) __bf16 v16bf;
typedef __attribute__((ext_vector_type(8)))  __bf16 v8bf;
typedef __attribute__((ext_vector_type(8)))  float  v8f;

#define NB 32          // batch
#define D_CH 256       // channels
#define NGRID 64
#define LDS_PITCH 272  // 256 + 16 bf16 pad to spread LDS banks

static __device__ __forceinline__ v8f wmma_bf16(v16bf a, v16bf b, v8f c) {
  return __builtin_amdgcn_wmma_f32_16x16x32_bf16(false, a, false, b, (short)0, c,
                                                 false, false);
}

static __device__ __forceinline__ v16bf load_a_frag(const __bf16* ap) {
  const int kh8 = ((threadIdx.x & 31) >> 4) * 8;
  v8bf lo = *(const v8bf*)(ap + kh8);      // K = kb + khalf*16 + 0..7
  v8bf hi = *(const v8bf*)(ap + kh8 + 8);  // K = kb + khalf*16 + 8..15
  return __builtin_shufflevector(lo, hi, 0,1,2,3,4,5,6,7,8,9,10,11,12,13,14,15);
}

// ---------------------------------------------------------------------------
// Projection GEMM: out[n][o] f32 = bias + W(256x256) * X.
// Block: 256 thr = 8 waves; wave w -> o-tiles {2w,2w+1}; 16 columns per block.
// ---------------------------------------------------------------------------
__global__ __launch_bounds__(256) void proj_gemm(
    const __bf16* __restrict__ W,      // [256][256]
    const __bf16* __restrict__ actIn,  // [Ncols][256]
    const float*  __restrict__ bias,   // [256] or nullptr
    float*  __restrict__ outF)         // [Ncols][256]
{
  const int lane  = threadIdx.x & 31;
  const int wave  = threadIdx.x >> 5;
  const int m     = lane & 15;
  const int khalf = lane >> 4;
  const int n     = blockIdx.x * 16 + m;
  const int ot0   = wave * 2;

  v8f acc0, acc1;
#pragma unroll
  for (int r = 0; r < 8; ++r) {
    float b0 = bias ? bias[(ot0    ) * 16 + khalf * 8 + r] : 0.0f;
    float b1 = bias ? bias[(ot0 + 1) * 16 + khalf * 8 + r] : 0.0f;
    acc0[r] = b0;
    acc1[r] = b1;
  }

  for (int kb = 0; kb < 256; kb += 32) {
    v16bf bvec = *(const v16bf*)(actIn + ((long)n << 8) + kb + khalf * 16);
    v16bf av0 = load_a_frag(W + (long)((ot0    ) * 16 + m) * 256 + kb + khalf * 8);
    v16bf av1 = load_a_frag(W + (long)((ot0 + 1) * 16 + m) * 256 + kb + khalf * 8);
    acc0 = wmma_bf16(av0, bvec, acc0);
    acc1 = wmma_bf16(av1, bvec, acc1);
  }

  const long base0 = ((long)n << 8) + (ot0    ) * 16 + khalf * 8;
  const long base1 = ((long)n << 8) + (ot0 + 1) * 16 + khalf * 8;
#pragma unroll
  for (int r = 0; r < 8; ++r) {
    outF[base0 + r] = acc0[r];
    outF[base1 + r] = acc1[r];
  }
}

// ---------------------------------------------------------------------------
// Fused conv stage + R projection.
//   Phase 1: act[n][o] = bias[o] + sum_{ktap,c} W[o][ktap*256+c] *
//            actIn[(b*Lin + t*cstride + ktap)][c],  n = b*Lout + t.
//            -> bf16 to global (next stage) and to LDS tile.
//   Phase 2: R[n][o] = sum_c Bm[o][c] * act[n][c]   (B-fragments from LDS).
// Block: 256 thr = 8 waves, 32 columns/block, wave tile = 2 o-tiles x 2 n-tiles
// (4 accumulators -> covers the 5-slot WMMA RAW hazard with independent WMMAs).
// Grid: Lout blocks (N = 32*Lout columns, exact).
// ---------------------------------------------------------------------------
__global__ __launch_bounds__(256) void fused_stage(
    const __bf16* __restrict__ W,      // [256][Ktaps*256]
    const __bf16* __restrict__ Bm,     // [256][256]
    const __bf16* __restrict__ actIn,  // [NB*Lin][256]
    const float*  __restrict__ bias,   // [256]
    __bf16* __restrict__ actOut,       // [NB*Lout][256]
    float*  __restrict__ Rout,         // [NB*Lout][256] (pre-offset into Rbig)
    int Lin, int Lout, int cstride, int Ktaps)
{
  __shared__ __bf16 tile[32 * LDS_PITCH];
  const int lane  = threadIdx.x & 31;
  const int wave  = threadIdx.x >> 5;
  const int m     = lane & 15;
  const int khalf = lane >> 4;
  const int ot0   = wave * 2;
  const int n0    = blockIdx.x * 32;
  const int nA    = n0 + m;        // n-tile 0 column
  const int nB    = n0 + 16 + m;   // n-tile 1 column
  const int bA = nA / Lout, tA = nA - bA * Lout;
  const int bB = nB / Lout, tB = nB - bB * Lout;
  const long srcA = (long)bA * Lin + (long)tA * cstride;
  const long srcB = (long)bB * Lin + (long)tB * cstride;
  const int Kdim  = Ktaps << 8;

  v8f acc00, acc01, acc10, acc11;  // [otile][ntile]
#pragma unroll
  for (int r = 0; r < 8; ++r) {
    float b0 = bias[(ot0    ) * 16 + khalf * 8 + r];
    float b1 = bias[(ot0 + 1) * 16 + khalf * 8 + r];
    acc00[r] = b0; acc01[r] = b0;
    acc10[r] = b1; acc11[r] = b1;
  }

  for (int kb = 0; kb < Kdim; kb += 32) {
    const int ktap = kb >> 8;
    const int coff = (kb & 255) + khalf * 16;
    v16bf bv0 = *(const v16bf*)(actIn + ((srcA + ktap) << 8) + coff);
    v16bf bv1 = *(const v16bf*)(actIn + ((srcB + ktap) << 8) + coff);
    v16bf av0 = load_a_frag(W + (long)((ot0    ) * 16 + m) * Kdim + kb + khalf * 8);
    v16bf av1 = load_a_frag(W + (long)((ot0 + 1) * 16 + m) * Kdim + kb + khalf * 8);
    acc00 = wmma_bf16(av0, bv0, acc00);
    acc01 = wmma_bf16(av0, bv1, acc01);
    acc10 = wmma_bf16(av1, bv0, acc10);
    acc11 = wmma_bf16(av1, bv1, acc11);
  }

  // D layout: VGPR r -> channel (otile*16 + khalf*8 + r), column n.
  const int ob0 = ot0 * 16 + khalf * 8;
  const int ob1 = ob0 + 16;
  v8bf s00, s01, s10, s11;
#pragma unroll
  for (int r = 0; r < 8; ++r) {
    s00[r] = (__bf16)acc00[r];
    s01[r] = (__bf16)acc01[r];
    s10[r] = (__bf16)acc10[r];
    s11[r] = (__bf16)acc11[r];
  }
  *(v8bf*)(actOut + ((long)nA << 8) + ob0) = s00;
  *(v8bf*)(actOut + ((long)nB << 8) + ob0) = s01;
  *(v8bf*)(actOut + ((long)nA << 8) + ob1) = s10;
  *(v8bf*)(actOut + ((long)nB << 8) + ob1) = s11;
  *(v8bf*)(&tile[(m     ) * LDS_PITCH + ob0]) = s00;
  *(v8bf*)(&tile[(16 + m) * LDS_PITCH + ob0]) = s01;
  *(v8bf*)(&tile[(m     ) * LDS_PITCH + ob1]) = s10;
  *(v8bf*)(&tile[(16 + m) * LDS_PITCH + ob1]) = s11;
  __syncthreads();

  // Phase 2: R = Bm * act  (K = 256, B from LDS)
  v8f r00, r01, r10, r11;
#pragma unroll
  for (int r = 0; r < 8; ++r) { r00[r] = 0.f; r01[r] = 0.f; r10[r] = 0.f; r11[r] = 0.f; }

  for (int kb = 0; kb < 256; kb += 32) {
    const int coff = kb + khalf * 16;
    v16bf bv0 = *(const v16bf*)(&tile[(m     ) * LDS_PITCH + coff]);
    v16bf bv1 = *(const v16bf*)(&tile[(16 + m) * LDS_PITCH + coff]);
    v16bf av0 = load_a_frag(Bm + (long)((ot0    ) * 16 + m) * 256 + kb + khalf * 8);
    v16bf av1 = load_a_frag(Bm + (long)((ot0 + 1) * 16 + m) * 256 + kb + khalf * 8);
    r00 = wmma_bf16(av0, bv0, r00);
    r01 = wmma_bf16(av0, bv1, r01);
    r10 = wmma_bf16(av1, bv0, r10);
    r11 = wmma_bf16(av1, bv1, r11);
  }

#pragma unroll
  for (int r = 0; r < 8; ++r) {
    Rout[((long)nA << 8) + ob0 + r] = r00[r];
    Rout[((long)nB << 8) + ob0 + r] = r01[r];
    Rout[((long)nA << 8) + ob1 + r] = r10[r];
    Rout[((long)nB << 8) + ob1 + r] = r11[r];
  }
}

// ---------------------------------------------------------------------------
// Pack kernels (bf16 conversion + layout)
// ---------------------------------------------------------------------------
__global__ __launch_bounds__(256) void pack_x(const float* __restrict__ x,
                                              __bf16* __restrict__ xbf) {
  int c = threadIdx.x, t = blockIdx.x, b = blockIdx.y;
  xbf[((b * NGRID + t) << 8) + c] = (__bf16)x[((b * D_CH) + c) * NGRID + t];
}

__global__ __launch_bounds__(256) void pack_w2(const float* __restrict__ w2,
                                               __bf16* __restrict__ Wbf2) {
  int c = threadIdx.x, o = blockIdx.x, s = blockIdx.y;
  long so = (long)s * 256 + o;
#pragma unroll
  for (int k = 0; k < 2; ++k)
    Wbf2[so * 512 + k * 256 + c] = (__bf16)w2[(so * 256 + c) * 2 + k];
}

__global__ __launch_bounds__(256) void pack_w3(const float* __restrict__ w3,
                                               __bf16* __restrict__ Wbf3) {
  int c = threadIdx.x, o = blockIdx.x, s = blockIdx.y;
  long so = (long)s * 256 + o;
#pragma unroll
  for (int k = 0; k < 3; ++k)
    Wbf3[so * 768 + k * 256 + c] = (__bf16)w3[(so * 256 + c) * 3 + k];
}

__global__ __launch_bounds__(256) void pack_vis(const float* __restrict__ vw,
                                                __bf16* __restrict__ Abf,
                                                __bf16* __restrict__ Bmbf,
                                                __bf16* __restrict__ Cmbf) {
  int c = threadIdx.x, o = blockIdx.x;
  const float* row = vw + (long)o * 768;
  Abf [o * 256 + c] = (__bf16)row[c];
  Bmbf[o * 256 + c] = (__bf16)row[256 + c];
  Cmbf[o * 256 + c] = (__bf16)row[512 + c];
}

// ---------------------------------------------------------------------------
// Final emitter: one block per (i, b). Builds per-row source table in LDS,
// then writes all 256 channels x 64 j's of out[b][:, i, :] fully coalesced.
//   populated j = off_s + i  iff  i % g_s == 0 and t = i/g_s < L_s
//   diag j = i
// ---------------------------------------------------------------------------
struct EmitTable {
  int off[32];
  int g[32];
  int L[32];
  int rowOff[32];   // row offset of stage s inside Rbig (units of 256 floats)
  int ns;
};

__global__ __launch_bounds__(256) void emit_out(
    const float* __restrict__ Pf,   // [NB*64][256]  (A*x + vis_b)
    const float* __restrict__ P2f,  // [NB*64][256]  (Bm*x)
    const float* __restrict__ Qf,   // [NB*64][256]  (Cm*x)
    const float* __restrict__ Rbig, // [32*sum(L)][256]
    const float* __restrict__ vb,   // [256]
    float* __restrict__ out,        // [NB][256][64][64]
    EmitTable tab)
{
  __shared__ int jsrc[64];   // -1 background, 0 diag, s+1 stage
  __shared__ int jRrow[64];  // row into Rbig for stage-populated j
  const int i = blockIdx.x, b = blockIdx.y, tid = threadIdx.x;

  if (tid < 64) jsrc[tid] = -1;
  __syncthreads();
  if (tid < tab.ns) {
    int g = tab.g[tid];
    if (i % g == 0) {
      int t = i / g;
      if (t < tab.L[tid]) {
        int j = tab.off[tid] + i;            // = off + t*g
        jsrc[j] = tid + 1;
        jRrow[j] = tab.rowOff[tid] + b * tab.L[tid] + t;
      }
    }
  }
  if (tid == 255) jsrc[i] = 0;               // diagonal (no stage has off==0)
  __syncthreads();

  const int jl = tid & 63;
  const int og = tid >> 6;                   // 0..3
  const long pi = ((long)b * NGRID + i) << 8;
  const int s = jsrc[jl];
  const long rrow = (s > 0) ? ((long)jRrow[jl] << 8) : 0;
  const long qj = (((long)b * NGRID + jl) << 8);

#pragma unroll 4
  for (int oo = 0; oo < 64; ++oo) {
    const int o = og * 64 + oo;
    float v;
    if (s < 0)       v = vb[o];
    else if (s == 0) v = Pf[pi + o] + P2f[pi + o] + Qf[pi + o];
    else             v = Pf[pi + o] + Rbig[rrow + o] + Qf[qj + o];
    out[(((long)b * 256 + o) * NGRID + i) * NGRID + jl] = v;
  }
}

// ---------------------------------------------------------------------------
// Host side
// ---------------------------------------------------------------------------
struct StageMeta { int ksz, cstride, offset, gstride, Lin, Lout; };

static int build_plan(StageMeta* st) {
  const int counts[3] = {15, 8, 8};
  int stride = 1, offset = 0, L = NGRID, idx = 0;
  for (int gi = 0; gi < 3; ++gi) {
    for (int k = 0; k < counts[gi]; ++k) {
      offset += stride;
      int ksz = (gi > 0 && k == 0) ? 3 : 2;
      int cs  = (ksz == 3) ? 2 : 1;
      int Lout = (ksz == 2) ? (L - 1) : ((L - 3) / 2 + 1);
      st[idx].ksz = ksz; st[idx].cstride = cs; st[idx].offset = offset;
      st[idx].gstride = stride; st[idx].Lin = L; st[idx].Lout = Lout;
      ++idx;
      L = Lout;
    }
    stride *= 2;
  }
  return idx; // 31
}

extern "C" void kernel_launch(void* const* d_in, const int* in_sizes, int n_in,
                              void* d_out, int out_size, void* d_ws, size_t ws_size,
                              hipStream_t stream) {
  const float* x  = (const float*)d_in[0];
  const float* w2 = (const float*)d_in[1];
  const float* b2 = (const float*)d_in[2];
  const float* w3 = (const float*)d_in[3];
  const float* b3 = (const float*)d_in[4];
  const float* vw = (const float*)d_in[5];
  const float* vb = (const float*)d_in[6];
  float* out = (float*)d_out;
  (void)in_sizes; (void)n_in; (void)out_size; (void)ws_size;

  // deterministic bump allocator on d_ws
  char* ws = (char*)d_ws;
  size_t off = 0;
  auto alloc = [&](size_t bytes) -> void* {
    void* p = ws + off;
    off = (off + bytes + 255) & ~(size_t)255;
    return p;
  };

  StageMeta st[31];
  int ns = build_plan(st);
  int rowOff[32];
  {
    int acc = 0;
    for (int s = 0; s < ns; ++s) { rowOff[s] = acc; acc += NB * st[s].Lout; }
    rowOff[ns] = acc;   // total rows = 33280
  }

  __bf16* xbf  = (__bf16*)alloc((size_t)NB * NGRID * 256 * 2);       // 1 MB
  __bf16* Wbf2 = (__bf16*)alloc((size_t)29 * 256 * 512 * 2);         // 7.6 MB
  __bf16* Wbf3 = (__bf16*)alloc((size_t)2 * 256 * 768 * 2);          // 0.75 MB
  __bf16* Abf  = (__bf16*)alloc((size_t)256 * 256 * 2);
  __bf16* Bmbf = (__bf16*)alloc((size_t)256 * 256 * 2);
  __bf16* Cmbf = (__bf16*)alloc((size_t)256 * 256 * 2);
  float*  Pf   = (float*)alloc((size_t)NB * NGRID * 256 * 4);        // 2 MB
  float*  P2f  = (float*)alloc((size_t)NB * NGRID * 256 * 4);
  float*  Qf   = (float*)alloc((size_t)NB * NGRID * 256 * 4);
  __bf16* actA = (__bf16*)alloc((size_t)NB * 63 * 256 * 2);          // ping
  __bf16* actB = (__bf16*)alloc((size_t)NB * 63 * 256 * 2);          // pong
  float*  Rbig = (float*)alloc((size_t)rowOff[ns] * 256 * 4);        // 34 MB

  // 1) pack inputs to bf16 GEMM layouts
  pack_x  <<<dim3(NGRID, NB), 256, 0, stream>>>(x, xbf);
  pack_w2 <<<dim3(256, 29),  256, 0, stream>>>(w2, Wbf2);
  pack_w3 <<<dim3(256, 2),   256, 0, stream>>>(w3, Wbf3);
  pack_vis<<<256,            256, 0, stream>>>(vw, Abf, Bmbf, Cmbf);

  // 2) projections of x: P = A*x + vis_b, P2 = Bm*x, Q = Cm*x  (2048 columns)
  proj_gemm<<<128, 256, 0, stream>>>(Abf,  xbf, vb,      Pf);
  proj_gemm<<<128, 256, 0, stream>>>(Bmbf, xbf, nullptr, P2f);
  proj_gemm<<<128, 256, 0, stream>>>(Cmbf, xbf, nullptr, Qf);

  // 3) fused conv + R chain (31 launches)
  const __bf16* cur = xbf;
  __bf16* nxt = actA;
  int i2 = 0, i3 = 0;
  for (int s = 0; s < ns; ++s) {
    const __bf16* W;
    const float* bias;
    int Kt;
    if (st[s].ksz == 2) { W = Wbf2 + (size_t)i2 * 256 * 512; bias = b2 + i2 * 256; Kt = 2; ++i2; }
    else                { W = Wbf3 + (size_t)i3 * 256 * 768; bias = b3 + i3 * 256; Kt = 3; ++i3; }

    fused_stage<<<st[s].Lout, 256, 0, stream>>>(
        W, Bmbf, cur, bias, nxt, Rbig + (size_t)rowOff[s] * 256,
        st[s].Lin, st[s].Lout, st[s].cstride, Kt);

    cur = nxt;
    nxt = (nxt == actA) ? actB : actA;
  }

  // 4) single coalesced emitter (background vis_b + diag + all scatters)
  EmitTable tab;
  for (int s = 0; s < ns; ++s) {
    tab.off[s] = st[s].offset;
    tab.g[s]   = st[s].gstride;
    tab.L[s]   = st[s].Lout;
    tab.rowOff[s] = rowOff[s];
  }
  for (int s = ns; s < 32; ++s) { tab.off[s] = tab.g[s] = tab.L[s] = tab.rowOff[s] = 0; }
  tab.ns = ns;
  emit_out<<<dim3(NGRID, NB), 256, 0, stream>>>(Pf, P2f, Qf, Rbig, vb, out, tab);
}
